// LSTMUnrolled_10471130268440
// MI455X (gfx1250) — matrix operand order repaired and checked by reading, not compiled, not run on hardware
//
#include <hip/hip_runtime.h>
#include <hip/hip_bf16.h>
#include <math.h>

typedef __attribute__((ext_vector_type(16))) __bf16   v16bf;
typedef __attribute__((ext_vector_type(8)))  float    v8f;
typedef __attribute__((ext_vector_type(4)))  float    v4f;
typedef __attribute__((ext_vector_type(8)))  unsigned v8u;

namespace {
constexpr int kB = 128;   // batch
constexpr int kT = 256;   // timesteps
constexpr int kD = 512;   // input dim
constexpr int kH = 128;   // hidden
constexpr int kG = 512;   // 4*H (== D)

// f32 -> bf16 via hardware convert (RNE); pack two into one dword.
__device__ __forceinline__ unsigned pack2_bf16(float lo, float hi) {
  unsigned short ua = __builtin_bit_cast(unsigned short, (__bf16)lo);
  unsigned short ub = __builtin_bit_cast(unsigned short, (__bf16)hi);
  return (unsigned)ua | ((unsigned)ub << 16);
}

__device__ __forceinline__ unsigned short f2bf(float x) {
  return __builtin_bit_cast(unsigned short, (__bf16)x);
}

__device__ __forceinline__ float sigmoidf_(float x) {
  return 1.0f / (1.0f + __expf(-x));
}
} // namespace

// ---------------------------------------------------------------------------
// Phase 1: G[t] = x[:,t,:] @ W_ih[t] + b_ih[t] + b_hh[t]   (time-parallel)
// Grid: (8 tiles, T). Block 256 thr = 8 waves; block tile 64(M) x 128(N),
// wave tile 32x32 (2x2 WMMA 16x16x32 bf16), K-loop over 512 in steps of 32.
// ---------------------------------------------------------------------------
__global__ __launch_bounds__(256)
void lstm_gates_gemm(const float* __restrict__ x,
                     const float* __restrict__ w_ih,
                     const float* __restrict__ b_ih,
                     const float* __restrict__ b_hh,
                     float* __restrict__ Gws)
{
  __shared__ __align__(16) unsigned short shA[64 * 34];   // bf16 [64 m][32 k] pad->34
  __shared__ __align__(16) unsigned short shB[128 * 34];  // bf16 [128 n][32 k] (transposed)

  const int t    = blockIdx.y;
  const int mblk = blockIdx.x & 1;     // 0..1 -> 64 rows each
  const int nblk = blockIdx.x >> 1;    // 0..3 -> 128 cols each
  const int m0 = mblk * 64;
  const int n0 = nblk * 128;

  const int tid  = threadIdx.x;
  const int wave = tid >> 5;
  const int lane = tid & 31;
  const int wy = wave >> 2;            // 0..1 : 32-row sub-tile
  const int wx = wave & 3;             // 0..3 : 32-col sub-tile
  const int ml = lane & 15;
  const int hh = lane >> 4;

  const float* wt = w_ih + (size_t)t * kD * kG;

  unsigned* shA32 = (unsigned*)shA;
  unsigned* shB32 = (unsigned*)shB;

  v8f acc[2][2] = {};

  for (int k0 = 0; k0 < kD; k0 += 32) {
    // ---- stage A: x rows m0..m0+63, cols k0..k0+31 (f32 -> bf16) ----
    #pragma unroll
    for (int rep = 0; rep < 2; ++rep) {
      int idx = tid + rep * 256;             // 512 float4s
      int row = idx >> 3;
      int kq  = idx & 7;
      const v4f f = __builtin_nontemporal_load(
          (const v4f*)(x + ((size_t)(m0 + row) * kT + t) * kD + k0 + kq * 4));
      shA32[row * 17 + kq * 2 + 0] = pack2_bf16(f.x, f.y);
      shA32[row * 17 + kq * 2 + 1] = pack2_bf16(f.z, f.w);
    }
    // ---- stage B transposed: W_ih rows k0..k0+31, cols n0..n0+127 ----
    #pragma unroll
    for (int rep = 0; rep < 4; ++rep) {
      int q  = tid + rep * 256;              // 1024 float4s
      int kk = q >> 5;
      int nq = (q & 31) * 4;
      const v4f f = __builtin_nontemporal_load(
          (const v4f*)(wt + (size_t)(k0 + kk) * kG + n0 + nq));
      shB[(nq + 0) * 34 + kk] = f2bf(f.x);
      shB[(nq + 1) * 34 + kk] = f2bf(f.y);
      shB[(nq + 2) * 34 + kk] = f2bf(f.z);
      shB[(nq + 3) * 34 + kk] = f2bf(f.w);
    }
    __syncthreads();

    // ---- fragments (CDNA5 16-bit A 16x32 / B 32x16 lane maps) ----
    v16bf afrag[2], bfrag[2];
    #pragma unroll
    for (int mt = 0; mt < 2; ++mt) {
      v8u u;
      const unsigned* p = shA32 + (wy * 32 + mt * 16 + ml) * 17;
      #pragma unroll
      for (int v = 0; v < 8; ++v) {
        int K = 2 * v + 16 * hh;   // j-pair -> K
        u[v] = p[K >> 1];
      }
      afrag[mt] = __builtin_bit_cast(v16bf, u);
    }
    #pragma unroll
    for (int nt = 0; nt < 2; ++nt) {
      v8u u;
      const unsigned* p = shB32 + (wx * 32 + nt * 16 + ml) * 17;
      #pragma unroll
      for (int v = 0; v < 8; ++v)
        u[v] = p[v + 8 * hh];                          // K = 2v + 16*hh
      bfrag[nt] = __builtin_bit_cast(v16bf, u);
    }
    #pragma unroll
    for (int mt = 0; mt < 2; ++mt)
      #pragma unroll
      for (int nt = 0; nt < 2; ++nt)
        acc[mt][nt] = __builtin_amdgcn_wmma_f32_16x16x32_bf16(
            false, afrag[mt], false, bfrag[nt], (short)0, acc[mt][nt], false, false);
    __syncthreads();
  }

  // ---- epilogue: + (b_ih + b_hh), store G[t] ----
  #pragma unroll
  for (int nt = 0; nt < 2; ++nt) {
    const int n_g = n0 + wx * 32 + nt * 16 + ml;
    const float bias = b_ih[t * kG + n_g] + b_hh[t * kG + n_g];
    #pragma unroll
    for (int mt = 0; mt < 2; ++mt) {
      #pragma unroll
      for (int r = 0; r < 8; ++r) {
        const int m_g = m0 + wy * 32 + mt * 16 + r + 8 * hh;   // D-tile row map
        Gws[((size_t)t * kB + m_g) * kG + n_g] = acc[mt][nt][r] + bias;
      }
    }
  }
}

// ---------------------------------------------------------------------------
// Phase 2: sequential scan. Batch rows are independent -> 8 blocks of 16 rows,
// NO inter-block sync. h-tile kept as bf16 in LDS; c lives in registers.
// Per step per wave: M-tile=16 rows, 4 gate N-tiles (its 16 hidden cols),
// K=128 in 4 WMMA steps -> 16 v_wmma per wave per timestep.
// ---------------------------------------------------------------------------
__global__ __launch_bounds__(256)
void lstm_scan(const float* __restrict__ w_hh,
               const float* __restrict__ Gws,
               float* __restrict__ out)
{
  __shared__ __align__(16) unsigned short shH[16 * 130];  // bf16 h tile, stride 130
  unsigned* shH32 = (unsigned*)shH;

  const int m0   = blockIdx.x * 16;     // this block's batch rows
  const int tid  = threadIdx.x;
  const int wave = tid >> 5;
  const int lane = tid & 31;
  const int ml = lane & 15;
  const int hh = lane >> 4;
  const int j0 = wave * 16;             // hidden cols owned by this wave

  for (int i = tid; i < 16 * 130; i += 256) shH[i] = 0;   // h_0 = 0
  __syncthreads();

  float c[8];
  #pragma unroll
  for (int r = 0; r < 8; ++r) c[r] = 0.0f;

  for (int t = 0; t < kT; ++t) {
    const float* Gt = Gws + ((size_t)t * kB + m0) * kG;
    const float* Wt = w_hh + (size_t)t * kG * kH;

    // init accumulators from precomputed gate pre-activations
    v8f acc[4];
    #pragma unroll
    for (int g = 0; g < 4; ++g) {
      const int n = g * kH + j0 + ml;
      #pragma unroll
      for (int r = 0; r < 8; ++r)
        acc[g][r] = Gt[(size_t)(r + 8 * hh) * kG + n];
    }

    // gates += h @ W_hh[t]^T  (B[k][n] = W_hh[n][k]: per-lane 16 contiguous K)
    #pragma unroll
    for (int kk = 0; kk < kH; kk += 32) {
      v8u ua;
      {
        const unsigned* p = shH32 + ml * 65 + (kk >> 1);
        #pragma unroll
        for (int v = 0; v < 8; ++v) {
          int K = 2 * v + 16 * hh;
          ua[v] = p[K >> 1];
        }
      }
      v16bf af = __builtin_bit_cast(v16bf, ua);
      #pragma unroll
      for (int g = 0; g < 4; ++g) {
        const float* wr = Wt + (size_t)(g * kH + j0 + ml) * kH + kk + 16 * hh;
        v4f f0 = *(const v4f*)(wr + 0);
        v4f f1 = *(const v4f*)(wr + 4);
        v4f f2 = *(const v4f*)(wr + 8);
        v4f f3 = *(const v4f*)(wr + 12);
        if (t + 1 < kT)  // pull next step's weights toward L2/L0
          __builtin_prefetch((const void*)(wr + (size_t)kG * kH), 0, 1);
        v8u ub;
        ub[0] = pack2_bf16(f0.x, f0.y); ub[1] = pack2_bf16(f0.z, f0.w);
        ub[2] = pack2_bf16(f1.x, f1.y); ub[3] = pack2_bf16(f1.z, f1.w);
        ub[4] = pack2_bf16(f2.x, f2.y); ub[5] = pack2_bf16(f2.z, f2.w);
        ub[6] = pack2_bf16(f3.x, f3.y); ub[7] = pack2_bf16(f3.z, f3.w);
        v16bf bf = __builtin_bit_cast(v16bf, ub);
        acc[g] = __builtin_amdgcn_wmma_f32_16x16x32_bf16(
            false, af, false, bf, (short)0, acc[g], false, false);
      }
    }
    __syncthreads();   // all LDS h reads for step t complete

    // elementwise cell update in D-fragment layout; c stays in registers
    #pragma unroll
    for (int r = 0; r < 8; ++r) {
      const float iv = sigmoidf_(acc[0][r]);
      const float fv = sigmoidf_(acc[1][r]);
      const float gv = tanhf(acc[2][r]);
      const float ov = sigmoidf_(acc[3][r]);
      c[r] = fv * c[r] + iv * gv;
      const float hv = ov * tanhf(c[r]);
      const int mr = r + 8 * hh;
      const int j  = j0 + ml;
      shH[mr * 130 + j] = f2bf(hv);
      out[((size_t)(m0 + mr) * kT + t) * kH + j] = hv;
    }
    __syncthreads();   // h(t) visible before next step's fragment reads
  }
}

// ---------------------------------------------------------------------------
extern "C" void kernel_launch(void* const* d_in, const int* in_sizes, int n_in,
                              void* d_out, int out_size, void* d_ws, size_t ws_size,
                              hipStream_t stream) {
  (void)in_sizes; (void)n_in; (void)out_size; (void)ws_size;
  const float* x    = (const float*)d_in[0];
  const float* w_ih = (const float*)d_in[1];
  const float* w_hh = (const float*)d_in[2];
  const float* b_ih = (const float*)d_in[3];
  const float* b_hh = (const float*)d_in[4];
  float* out = (float*)d_out;
  float* Gws = (float*)d_ws;   // T*B*4H f32 = 64 MiB of scratch (fits in L2)

  // Phase 1: 256 t x (2 M-blocks x 4 N-blocks) = 2048 workgroups
  lstm_gates_gemm<<<dim3(8, kT), 256, 0, stream>>>(x, w_ih, b_ih, b_hh, Gws);
  // Phase 2: 8 independent batch-row blocks, sequential over T internally
  lstm_scan<<<8, 256, 0, stream>>>(w_hh, Gws, out);
}
